// MaskFlowNet_v2_893353197636
// MI455X (gfx1250) — matrix operand order repaired and checked by reading, not compiled, not run on hardware
//
#include <hip/hip_runtime.h>

typedef _Float16 f16;
typedef __attribute__((ext_vector_type(16))) _Float16 v16h;
typedef __attribute__((ext_vector_type(8)))  float    v8f;

#define LRELU(v) ((v) >= 0.f ? (v) : 0.1f * (v))

// ---------------------------------------------------------------------------
// Weight packing: f32 OIHW -> f16 WMMA A-fragment layout.
// Block (ct,kt) holds a 16(cout) x 32(k) tile as [lane 0..31][half 0..15],
// in contiguous-K order (k = 16*(lane/16) + half), so the GEMM kernel loads
// one contiguous 32B v16h per lane per K-step.
// ---------------------------------------------------------------------------
__global__ void k_pack3x3(const float* __restrict__ w, f16* __restrict__ pk,
                          int Cout, int K, int Ct, int Kt) {
  int idx = blockIdx.x * 256 + threadIdx.x;
  int total = Ct * Kt * 512;
  if (idx >= total) return;
  int i = idx & 15;
  int lane = (idx >> 4) & 31;
  int blk = idx >> 9;
  int kt = blk % Kt, ct = blk / Kt;
  int v = i >> 1, h = i & 1;
  int kk = (lane >> 4) * 16 + i;
  int m = lane & 15;
  int co = ct * 16 + m;
  int k = kt * 32 + kk;
  float val = (co < Cout && k < K) ? w[(size_t)co * K + k] : 0.f;
  pk[idx] = (f16)val;
}

// ---------------------------------------------------------------------------
// Implicit-GEMM 3x3 conv via V_WMMA_F32_16X16X32_F16, M-blocked by MB.
// One wave computes a (16*MB cout) x 16(pixel) tile; K = Cin*9 in steps of 32.
// B fragment is assembled branch-free in registers (clamped unconditional
// loads + cndmask zeroing, incremental (ci,ky,kx) decode); the gather is
// amortized over MB back-to-back WMMAs sharing it.
// B layout here (same K order as the A packing): lane<16 owns (pixel=lane, K=0..15),
// lane>=16 owns (pixel=lane-16, K=16..31) -> contiguous-K per lane, no LDS.
// ---------------------------------------------------------------------------
template <int MB>
__global__ void k_conv3x3_wmma(const float* __restrict__ x,
                               const f16* __restrict__ wpk,
                               const float* __restrict__ bias,
                               float* __restrict__ y,
                               int Nb, int Cin, int H, int W, int Cout,
                               int Ho, int Wo, int stride, int pad, int dil,
                               int act, int Kt) {
  const int lane = threadIdx.x & 31;
  const int wid = threadIdx.x >> 5;
  const int HWo = Ho * Wo;
  const int P = Nb * HWo;
  const int tiles_p = (P + 15) >> 4;
  const int tiles_m = (Cout + 15) >> 4;
  const int mg = (tiles_m + MB - 1) / MB;
  const int maxTile = tiles_p * mg - 1;
  int tile = blockIdx.x * 8 + wid;
  int tc = tile > maxTile ? maxTile : tile;
  const int tg = tc / tiles_p;
  const int tp = tc - tg * tiles_p;
  int pm = tp * 16 + (lane & 15);
  if (pm >= P) pm = P - 1;
  const int n = pm / HWo;
  const int rem = pm - n * HWo;
  const int oy = rem / Wo;
  const int ox = rem - oy * Wo;
  const int iy0 = oy * stride - pad;
  const int ix0 = ox * stride - pad;
  const int K = Cin * 9;
  const int khalf = (lane >> 4) * 16;
  const int HWi = H * W;
  const float* __restrict__ xn = x + (size_t)n * Cin * HWi;
  v8f acc[MB];
#pragma unroll
  for (int m = 0; m < MB; ++m) acc[m] = (v8f){};
  for (int kt = 0; kt < Kt; ++kt) {
    // A fragments: pre-packed, one contiguous 32B load per lane per sub-tile
    v16h a[MB];
#pragma unroll
    for (int m = 0; m < MB; ++m) {
      int tmi = tg * MB + m;
      if (tmi > tiles_m - 1) tmi = tiles_m - 1;
      a[m] = *(const v16h*)(wpk + (size_t)(tmi * Kt + kt) * 512 + lane * 16);
    }
    // B fragment: branch-free implicit im2col, decoded incrementally
    int k0 = kt * 32 + khalf;
    int ci = k0 / 9;
    int r = k0 - ci * 9;
    int ky = r / 3;
    int kx = r - ky * 3;
    v16h b;
#pragma unroll
    for (int j = 0; j < 16; ++j) {
      int iy = iy0 + ky * dil;
      int ix = ix0 + kx * dil;
      bool ok = (k0 + j < K) & ((unsigned)iy < (unsigned)H) &
                ((unsigned)ix < (unsigned)W);
      int cic = ci < Cin - 1 ? ci : Cin - 1;
      int iyc = iy < 0 ? 0 : (iy > H - 1 ? H - 1 : iy);
      int ixc = ix < 0 ? 0 : (ix > W - 1 ? W - 1 : ix);
      float ld = xn[(size_t)cic * HWi + (size_t)iyc * W + ixc];
      b[j] = (f16)(ok ? ld : 0.f);
      // advance (ci, ky, kx)
      ++kx;
      if (kx == 3) {
        kx = 0;
        ++ky;
        if (ky == 3) {
          ky = 0;
          ++ci;
        }
      }
    }
#pragma unroll
    for (int m = 0; m < MB; ++m)
      acc[m] = __builtin_amdgcn_wmma_f32_16x16x32_f16(false, a[m], false, b,
                                                      (short)0, acc[m], false,
                                                      false);
  }
  if (tile <= maxTile) {
    int p = tp * 16 + (lane & 15);
    if (p < P) {
      int n2 = p / HWo;
      int r2 = p - n2 * HWo;
      int mh = (lane >= 16) ? 8 : 0;
#pragma unroll
      for (int m = 0; m < MB; ++m) {
        int tmi = tg * MB + m;
        if (tmi < tiles_m) {
#pragma unroll
          for (int r = 0; r < 8; ++r) {
            int co = tmi * 16 + r + mh;
            if (co < Cout) {
              float v = acc[m][r] + bias[co];
              if (act) v = LRELU(v);
              y[((size_t)n2 * Cout + co) * HWo + r2] = v;
            }
          }
        }
      }
    }
  }
}

// ---------------------------------------------------------------------------
// ConvTranspose2d(k=4,s=2,p=1) as written in the reference (lhs_dilation=2).
// w layout: (O=16, I=Cin, 4, 4). Output (2H, 2W). Cheap vs the WMMA convs.
// ---------------------------------------------------------------------------
__global__ void k_deconv4x4(const float* __restrict__ x,
                            const float* __restrict__ w,
                            const float* __restrict__ b, float* __restrict__ y,
                            int Nb, int Cin, int H, int W, int Cout) {
  int Ho = 2 * H, Wo = 2 * W;
  int total = Nb * Cout * Ho * Wo;
  int i = blockIdx.x * 256 + threadIdx.x;
  if (i >= total) return;
  int xx = i % Wo;
  int t = i / Wo;
  int yy = t % Ho;
  t /= Ho;
  int co = t % Cout;
  int n = t / Cout;
  float acc = b[co];
  for (int ky = 0; ky < 4; ++ky) {
    int sy = yy - 2 + ky;
    if (sy & 1) continue;
    sy >>= 1;
    if ((unsigned)sy >= (unsigned)H) continue;
    for (int kx = 0; kx < 4; ++kx) {
      int sx = xx - 2 + kx;
      if (sx & 1) continue;
      sx >>= 1;
      if ((unsigned)sx >= (unsigned)W) continue;
      const float* xp = x + ((size_t)n * Cin * H + sy) * W + sx;
      const float* wp = w + ((size_t)co * Cin * 16) + (ky * 4 + kx);
      for (int ci = 0; ci < Cin; ++ci)
        acc += xp[(size_t)ci * H * W] * wp[(size_t)ci * 16];
    }
  }
  y[i] = acc;
}

// Bilinear backward warp, fused optional sigmoid(mask) scale + leaky ReLU.
__global__ void k_warp(const float* __restrict__ x,
                       const float* __restrict__ flow,
                       const float* __restrict__ mask, float* __restrict__ y,
                       int Nb, int C, int H, int W) {
  int total = Nb * C * H * W;
  int i = blockIdx.x * 256 + threadIdx.x;
  if (i >= total) return;
  int xx = i % W;
  int t = i / W;
  int yy = t % H;
  t /= H;
  int c = t % C;
  int n = t / C;
  size_t HW = (size_t)H * W;
  size_t fb = (size_t)n * 2 * HW + (size_t)yy * W + xx;
  float cx = (float)xx + flow[fb];
  float cy = (float)yy + flow[fb + HW];
  float x0 = floorf(cx), y0 = floorf(cy);
  int xi = (int)x0, yi = (int)y0;
  float wx = cx - x0, wy = cy - y0;
  const float* b = x + ((size_t)n * C + c) * HW;
  auto g = [&](int r, int q) -> float {
    bool ok = (r >= 0) & (r <= H - 1) & (q >= 0) & (q <= W - 1);
    int rc = r < 0 ? 0 : (r > H - 1 ? H - 1 : r);
    int qc = q < 0 ? 0 : (q > W - 1 ? W - 1 : q);
    float ld = b[(size_t)rc * W + qc];
    return ok ? ld : 0.f;
  };
  float v = g(yi, xi) * (1 - wy) * (1 - wx) + g(yi, xi + 1) * (1 - wy) * wx +
            g(yi + 1, xi) * wy * (1 - wx) + g(yi + 1, xi + 1) * wy * wx;
  if (mask) {
    float m = mask[(size_t)n * HW + (size_t)yy * W + xx];
    v *= 1.f / (1.f + expf(-m));
  }
  y[i] = LRELU(v);
}

// 81-displacement cost volume (mean over C), fused leaky ReLU.
__global__ void k_corr(const float* __restrict__ f1, const float* __restrict__ f2,
                       float* __restrict__ y, int Nb, int C, int H, int W) {
  int total = Nb * 81 * H * W;
  int i = blockIdx.x * 256 + threadIdx.x;
  if (i >= total) return;
  int xx = i % W;
  int t = i / W;
  int yy = t % H;
  t /= H;
  int d = t % 81;
  int n = t / 81;
  int dy = d / 9 - 4;
  int dx = d - (d / 9) * 9 - 4;
  int sy = yy + dy, sx = xx + dx;
  float acc = 0.f;
  if ((unsigned)sy < (unsigned)H && (unsigned)sx < (unsigned)W) {
    size_t HW = (size_t)H * W;
    const float* p1 = f1 + (size_t)n * C * HW + (size_t)yy * W + xx;
    const float* p2 = f2 + (size_t)n * C * HW + (size_t)sy * W + sx;
    for (int c = 0; c < C; ++c) acc += p1[(size_t)c * HW] * p2[(size_t)c * HW];
  }
  acc *= (1.f / (float)C);
  y[i] = LRELU(acc);
}

// 2x bilinear upsample, align_corners=True, with output scale (2.0 for flow).
__global__ void k_up2(const float* __restrict__ x, float* __restrict__ y,
                      int Nb, int C, int H, int W, float scale) {
  int Ho = 2 * H, Wo = 2 * W;
  int total = Nb * C * Ho * Wo;
  int i = blockIdx.x * 256 + threadIdx.x;
  if (i >= total) return;
  int xx = i % Wo;
  int t = i / Wo;
  int yy = t % Ho;
  t /= Ho;
  int c = t % C;
  int n = t / C;
  float py = (float)yy * (float)(H - 1) / (float)(Ho - 1);
  float px = (float)xx * (float)(W - 1) / (float)(Wo - 1);
  int i0 = (int)py;
  if (i0 > H - 2) i0 = H - 2;
  int j0 = (int)px;
  if (j0 > W - 2) j0 = W - 2;
  float ty = py - i0, tx = px - j0;
  const float* b = x + ((size_t)n * C + c) * H * W;
  float v = b[(size_t)i0 * W + j0] * (1 - ty) * (1 - tx) +
            b[(size_t)i0 * W + j0 + 1] * (1 - ty) * tx +
            b[(size_t)(i0 + 1) * W + j0] * ty * (1 - tx) +
            b[(size_t)(i0 + 1) * W + j0 + 1] * ty * tx;
  y[i] = v * scale;
}

__global__ void k_fill(float* p, float v, int n) {
  int i = blockIdx.x * 256 + threadIdx.x;
  if (i < n) p[i] = v;
}

__global__ void k_add(float* d, const float* s, int n) {
  int i = blockIdx.x * 256 + threadIdx.x;
  if (i < n) d[i] += s[i];
}

// Copy [Nb,Cs,HW] tensor into channel slot `co` of concat buffer [Nb,Ct,HW].
__global__ void k_copy_ch(const float* __restrict__ s, float* __restrict__ d,
                          int Nb, int Cs, int HW, int Ct, int co) {
  int i = blockIdx.x * 256 + threadIdx.x;
  int total = Nb * Cs * HW;
  if (i >= total) return;
  int hw = i % HW;
  int t = i / HW;
  int c = t % Cs;
  int n = t / Cs;
  d[((size_t)n * Ct + co + c) * HW + hw] = s[i];
}

// ---------------------------------------------------------------------------
// Host orchestration
// ---------------------------------------------------------------------------
struct PK {
  f16* pk;
  const float* b;
  int Cin, Cout, Kt;
};

extern "C" void kernel_launch(void* const* d_in, const int* in_sizes, int n_in,
                              void* d_out, int out_size, void* d_ws,
                              size_t ws_size, hipStream_t stream) {
  (void)in_sizes; (void)n_in; (void)out_size; (void)ws_size;
  float* out = (float*)d_out;

  // ---- parameter unpacking (jax pytree order: sorted dict keys) ----
  // top dict: x1_raw, x2_raw, params{cnet,fe,fms}; leaf dicts {'b','w'} -> b,w
  int pi = 0;
  const float* x1 = (const float*)d_in[pi++];
  const float* x2 = (const float*)d_in[pi++];
  const float *cnetB[5][7], *cnetW[5][7];
  for (int l = 0; l < 5; ++l)
    for (int j = 0; j < 7; ++j) {
      cnetB[l][j] = (const float*)d_in[pi++];
      cnetW[l][j] = (const float*)d_in[pi++];
    }
  const float *feB[6][2], *feW[6][2];
  for (int l = 0; l < 6; ++l)
    for (int j = 0; j < 2; ++j) {
      feB[l][j] = (const float*)d_in[pi++];
      feW[l][j] = (const float*)d_in[pi++];
    }
  // fms level keys sorted: c0..c4, flow, mask, up
  const float *fmsB[5][8], *fmsW[5][8];
  for (int l = 0; l < 5; ++l)
    for (int j = 0; j < 8; ++j) {
      fmsB[l][j] = (const float*)d_in[pi++];
      fmsW[l][j] = (const float*)d_in[pi++];
    }

  // ---- workspace bump allocator ----
  char* base = (char*)d_ws;
  size_t off = 0;
  auto allocB = [&](size_t bytes) -> void* {
    void* p = base + off;
    off += (bytes + 255) & ~(size_t)255;
    return p;
  };
  auto allocF = [&](size_t n) -> float* { return (float*)allocB(n * 4); };
  auto G = [](long total) { return (int)((total + 255) / 256); };

  // ---- pack all 3x3 conv weights to f16 WMMA fragments ----
  auto mkpk = [&](const float* w, const float* b, int Cin, int Cout) -> PK {
    int K = Cin * 9, Kt = (K + 31) / 32, Ct = (Cout + 15) / 16;
    size_t halfs = (size_t)Ct * Kt * 512;
    f16* pk = (f16*)allocB(halfs * sizeof(f16));
    k_pack3x3<<<G((long)halfs), 256, 0, stream>>>(w, pk, Cout, K, Ct, Kt);
    return {pk, b, Cin, Cout, Kt};
  };

  const int feCh[7] = {3, 16, 32, 64, 96, 128, 196};
  PK fePK[6][2];
  for (int l = 0; l < 6; ++l) {
    fePK[l][0] = mkpk(feW[l][0], feB[l][0], feCh[l], feCh[l + 1]);
    fePK[l][1] = mkpk(feW[l][1], feB[l][1], feCh[l + 1], feCh[l + 1]);
  }
  const int cnetIn[5] = {215, 147, 115, 83, 51};
  const int cnetWd[6] = {128, 128, 128, 96, 64, 32};
  PK cnetPK[5][7];
  for (int l = 0; l < 5; ++l) {
    int c = cnetIn[l];
    for (int j = 0; j < 6; ++j) {
      cnetPK[l][j] = mkpk(cnetW[l][j], cnetB[l][j], c, cnetWd[j]);
      c = cnetWd[j];
    }
    cnetPK[l][6] = mkpk(cnetW[l][6], cnetB[l][6], 32, 2);
  }
  const int fmsIn[5] = {296, 228, 196, 164, 132};
  const int fmsWd[5] = {128, 128, 96, 64, 32};
  PK fmsPK[5][7];  // c0..c4, flow, mask
  for (int l = 0; l < 5; ++l) {
    int c = fmsIn[l];
    for (int j = 0; j < 5; ++j) {
      fmsPK[l][j] = mkpk(fmsW[l][j], fmsB[l][j], c, fmsWd[j]);
      c = fmsWd[j];
    }
    fmsPK[l][5] = mkpk(fmsW[l][5], fmsB[l][5], 32, 2);
    fmsPK[l][6] = mkpk(fmsW[l][6], fmsB[l][6], 32, 1);
  }

  auto runConv = [&](const PK& p, const float* x, float* y, int H, int W,
                     int Ho, int Wo, int stride, int pad, int dil, int act) {
    long Pn = 2L * Ho * Wo;
    int tiles_m = (p.Cout + 15) / 16;
    int MB = tiles_m >= 4 ? 4 : 1;
    long tiles = (long)((tiles_m + MB - 1) / MB) * ((Pn + 15) / 16);
    int blocks = (int)((tiles + 7) / 8);
    if (MB == 4)
      k_conv3x3_wmma<4><<<blocks, 256, 0, stream>>>(
          x, p.pk, p.b, y, 2, p.Cin, H, W, p.Cout, Ho, Wo, stride, pad, dil,
          act, p.Kt);
    else
      k_conv3x3_wmma<1><<<blocks, 256, 0, stream>>>(
          x, p.pk, p.b, y, 2, p.Cin, H, W, p.Cout, Ho, Wo, stride, pad, dil,
          act, p.Kt);
  };

  // ---- feature pyramid ----
  const int chans[6] = {16, 32, 64, 96, 128, 196};
  const int Hs[6] = {192, 96, 48, 24, 12, 6};
  const int Ws[6] = {384, 192, 96, 48, 24, 12};
  float* tmp = allocF((size_t)2 * 16 * 192 * 384);
  float* f1[6];
  float* f2[6];
  for (int l = 0; l < 6; ++l) {
    f1[l] = allocF((size_t)2 * chans[l] * Hs[l] * Ws[l]);
    f2[l] = allocF((size_t)2 * chans[l] * Hs[l] * Ws[l]);
  }
  for (int img = 0; img < 2; ++img) {
    const float* cur = img ? x2 : x1;
    int Hin = 384, Win = 768;
    for (int l = 0; l < 6; ++l) {
      runConv(fePK[l][0], cur, tmp, Hin, Win, Hs[l], Ws[l], 2, 1, 1, 1);
      float* fo = (img ? f2 : f1)[l];
      runConv(fePK[l][1], tmp, fo, Hs[l], Ws[l], Hs[l], Ws[l], 1, 1, 1, 1);
      cur = fo;
      Hin = Hs[l];
      Win = Ws[l];
    }
  }

  // ---- decoder scratch (sized for the largest processed level, lv=1) ----
  const int maxHW = 96 * 192;
  float* flowPrev = allocF((size_t)2 * 2 * maxHW);
  float* flowCur = allocF((size_t)2 * 2 * maxHW);
  float* maskPrev = allocF((size_t)2 * maxHW);
  float* maskCur = allocF((size_t)2 * maxHW);
  float* upfIn = allocF((size_t)2 * 16 * 4 * maxHW);   // deconv doubles H,W
  float* upfNext = allocF((size_t)2 * 16 * 4 * maxHW);
  float* wrp = allocF((size_t)2 * 32 * maxHW);
  float* corr = allocF((size_t)2 * 81 * maxHW);
  float* bufX = allocF((size_t)2 * 132 * maxHW);
  float* bufCC = allocF((size_t)2 * 51 * maxHW);
  float* t0 = allocF((size_t)2 * 128 * maxHW);
  float* t1 = allocF((size_t)2 * 128 * maxHW);
  float* bufFC = allocF((size_t)2 * 2 * maxHW);
  float* bufFF = allocF((size_t)2 * 2 * maxHW);

  const size_t outOff[5] = {0, 288, 1440, 6048, 24480};
  const int cnetDil[6] = {1, 2, 4, 8, 16, 1};

  for (int idx = 0; idx < 5; ++idx) {
    int lv = 5 - idx;
    int F = chans[lv], H = Hs[lv], W = Ws[lv], HW = H * W;
    if (idx == 0) {
      k_fill<<<G(2L * 2 * HW), 256, 0, stream>>>(flowCur, 0.f, 2 * 2 * HW);
      k_fill<<<G(2L * HW), 256, 0, stream>>>(maskCur, 1.f, 2 * HW);
      k_fill<<<G(2L * 16 * HW), 256, 0, stream>>>(upfIn, 0.f, 2 * 16 * HW);
    } else {
      k_up2<<<G(2L * 2 * HW), 256, 0, stream>>>(flowPrev, flowCur, 2, 2, H / 2,
                                                W / 2, 2.f);
      k_up2<<<G(2L * HW), 256, 0, stream>>>(maskPrev, maskCur, 2, 1, H / 2,
                                            W / 2, 1.f);
    }
    // warp f2 by flow (sigmoid(mask) scale only for idx>0), fused lrelu
    k_warp<<<G(2L * F * HW), 256, 0, stream>>>(
        f2[lv], flowCur, idx == 0 ? (const float*)nullptr : maskCur, wrp, 2, F,
        H, W);
    // 81-channel cost volume, fused lrelu
    k_corr<<<G(2L * 81 * HW), 256, 0, stream>>>(f1[lv], wrp, corr, 2, F, H, W);
    // concat [corr, x1, upf, flow, mask] -> C = F + 100
    int Ct = F + 100;
    k_copy_ch<<<G(2L * 81 * HW), 256, 0, stream>>>(corr, bufX, 2, 81, HW, Ct, 0);
    k_copy_ch<<<G(2L * F * HW), 256, 0, stream>>>(f1[lv], bufX, 2, F, HW, Ct, 81);
    k_copy_ch<<<G(2L * 16 * HW), 256, 0, stream>>>(upfIn, bufX, 2, 16, HW, Ct,
                                                   81 + F);
    k_copy_ch<<<G(2L * 2 * HW), 256, 0, stream>>>(flowCur, bufX, 2, 2, HW, Ct,
                                                  97 + F);
    k_copy_ch<<<G(2L * HW), 256, 0, stream>>>(maskCur, bufX, 2, 1, HW, Ct,
                                              99 + F);
    // flow estimator trunk: 5 WMMA convs with lrelu
    const float* cin = bufX;
    float* pp[2] = {t0, t1};
    for (int j = 0; j < 5; ++j) {
      runConv(fmsPK[idx][j], cin, pp[j & 1], H, W, H, W, 1, 1, 1, 1);
      cin = pp[j & 1];
    }
    const float* x32 = t0;  // after 5 convs result lands in t0
    if (idx < 4)
      k_deconv4x4<<<G(2L * 16 * 4 * HW), 256, 0, stream>>>(
          x32, fmsW[idx][7], fmsB[idx][7], upfNext, 2, 32, H, W, 16);
    runConv(fmsPK[idx][5], x32, bufFC, H, W, H, W, 1, 1, 1, 0);  // flow head
    k_add<<<G(2L * 2 * HW), 256, 0, stream>>>(bufFC, flowCur, 2 * 2 * HW);
    runConv(fmsPK[idx][6], x32, maskPrev, H, W, H, W, 1, 1, 1, 0);  // new mask
    // context network on [x1, upf_in, flow_in, new mask] -> C = F + 19
    int Ct2 = F + 19;
    k_copy_ch<<<G(2L * F * HW), 256, 0, stream>>>(f1[lv], bufCC, 2, F, HW, Ct2, 0);
    k_copy_ch<<<G(2L * 16 * HW), 256, 0, stream>>>(upfIn, bufCC, 2, 16, HW, Ct2, F);
    k_copy_ch<<<G(2L * 2 * HW), 256, 0, stream>>>(flowCur, bufCC, 2, 2, HW, Ct2,
                                                  F + 16);
    k_copy_ch<<<G(2L * HW), 256, 0, stream>>>(maskPrev, bufCC, 2, 1, HW, Ct2,
                                              F + 18);
    cin = bufCC;
    for (int j = 0; j < 6; ++j) {
      float* o = (j & 1) ? t0 : t1;
      runConv(cnetPK[idx][j], cin, o, H, W, H, W, 1, cnetDil[j], cnetDil[j], 1);
      cin = o;
    }
    runConv(cnetPK[idx][6], cin, bufFF, H, W, H, W, 1, 1, 1, 0);
    k_add<<<G(2L * 2 * HW), 256, 0, stream>>>(bufFC, bufFF, 2 * 2 * HW);
    // emit flow for this level
    if (idx < 4) {
      hipMemcpyAsync(out + outOff[idx], bufFC, (size_t)2 * 2 * HW * 4,
                     hipMemcpyDeviceToDevice, stream);
      hipMemcpyAsync(flowPrev, bufFC, (size_t)2 * 2 * HW * 4,
                     hipMemcpyDeviceToDevice, stream);
      float* sw = upfIn;
      upfIn = upfNext;
      upfNext = sw;
    } else {
      // final: flows[-1] = up2(flow2) * 2
      k_up2<<<G(2L * 2 * 4 * HW), 256, 0, stream>>>(bufFC, out + outOff[4], 2,
                                                    2, H, W, 2.f);
    }
  }
}
